// SURGE_79671643341679
// MI455X (gfx1250) — matrix operand order repaired and checked by reading, not compiled, not run on hardware
//
#include <hip/hip_runtime.h>

#define NEG 0.2f
#define GNUM 1024
#define NPG 64
#define NMOL 54
#define NNODES 65536
#define NEDGE 393216
#define EEDGE (NEDGE + NNODES)

typedef __attribute__((ext_vector_type(16))) _Float16 v16h;
typedef __attribute__((ext_vector_type(8)))  _Float16 v8h;
typedef __attribute__((ext_vector_type(8)))  float    v8f;

__device__ __forceinline__ float lrelu(float v) { return v > 0.f ? v : NEG * v; }

__device__ __forceinline__ void atomicMaxF(float* addr, float val) {
    if (val >= 0.f) atomicMax((int*)addr, __float_as_int(val));
    else            atomicMin((unsigned int*)addr, (unsigned int)__float_as_int(val));
}

// ---------------------------------------------------------------------------
// f32 -> padded f16 conversion:  Xh[r, k] = (k < K) ? X[r,k] : 0   (Kp mult of 32)
// ---------------------------------------------------------------------------
__global__ void k_cvt_pad_f16(const float* __restrict__ X, _Float16* __restrict__ Xh,
                              long rows, int K, int Kp) {
    long i = (long)blockIdx.x * blockDim.x + threadIdx.x;
    if (i >= rows * Kp) return;
    long r = i / Kp;
    int  k = (int)(i % Kp);
    Xh[i] = (k < K) ? (_Float16)X[r * K + k] : (_Float16)0.f;
}

// ---------------------------------------------------------------------------
// Pack W[K,N] (f32) into per-lane WMMA B-fragment order (f16):
//   Wp[kt][nt][lane][e] = W[(kt*32 + (e&7) + 16*(e>>3) + 8*(lane>=16)) * N + nt*16 + (lane&15)]
// so the GEMM loads one contiguous v16h (32B) per lane per K-step.
// ---------------------------------------------------------------------------
__global__ void k_pack_w(const float* __restrict__ W, _Float16* __restrict__ Wp,
                         int K, int Kp, int N) {
    int total = (Kp / 32) * (N / 16) * 512;
    int i = blockIdx.x * blockDim.x + threadIdx.x;
    if (i >= total) return;
    int e    = i & 15;
    int lane = (i >> 4) & 31;
    int tile = i >> 9;              // kt * (N/16) + nt
    int ntiles = N / 16;
    int kt = tile / ntiles, nt = tile % ntiles;
    int half = lane >> 4, l16 = lane & 15;
    int kb = kt * 32 + (e & 7) + ((e >> 3) << 4) + (half << 3);
    int c  = nt * 16 + l16;
    Wp[i] = (kb < K) ? (_Float16)W[(size_t)kb * N + c] : (_Float16)0.f;
}

// ---------------------------------------------------------------------------
// WMMA GEMM: Y[M,N] = Xh[M,Kp](f16) @ Wp(packed f16) + bias[N]
// Each wave computes a 64x16 output strip (4 independent 16x16 accumulators):
// per K-step it loads B once (one v16h) and 4 A fragments, issuing 4
// independent v_wmma_f32_16x16x32_f16 — B reuse 4x, and the independent
// accumulators hide WMMA->WMMA latency. 4 waves/block => 256 rows per block.
// ---------------------------------------------------------------------------
__global__ void k_gemm_wmma(const _Float16* __restrict__ Xh, const _Float16* __restrict__ Wp,
                            const float* __restrict__ bias, float* __restrict__ Y,
                            int M, int Kp, int N) {
    int wave = threadIdx.x >> 5;
    int lane = threadIdx.x & 31;
    int tn = blockIdx.y;
    int row0 = (blockIdx.x * 4 + wave) * 64;     // 4 row tiles of 16
    if (row0 >= M) return;
    int half = lane >> 4;
    int l16  = lane & 15;
    int c = tn * 16 + l16;

    const _Float16* a0 = Xh + (size_t)(row0 + l16) * Kp + (half << 3);
    size_t rstride = (size_t)16 * Kp;            // 16 rows of A
    const _Float16* bptr = Wp + ((size_t)tn << 9) + (lane << 4);
    size_t bstep = ((size_t)(N >> 4)) << 9;      // advance one kt

    v8f acc0 = {0.f,0.f,0.f,0.f,0.f,0.f,0.f,0.f};
    v8f acc1 = acc0, acc2 = acc0, acc3 = acc0;
    int nkt = Kp >> 5;
    for (int kt = 0; kt < nkt; ++kt) {
        v16h b = *(const v16h*)(bptr);
        bptr += bstep;
        const _Float16* ak = a0 + (kt << 5);
        if (kt + 1 < nkt) __builtin_prefetch(ak + 32, 0, 1);   // global_prefetch_b8

        v8h lo0 = *(const v8h*)(ak);
        v8h hi0 = *(const v8h*)(ak + 16);
        v8h lo1 = *(const v8h*)(ak + rstride);
        v8h hi1 = *(const v8h*)(ak + rstride + 16);
        v8h lo2 = *(const v8h*)(ak + 2 * rstride);
        v8h hi2 = *(const v8h*)(ak + 2 * rstride + 16);
        v8h lo3 = *(const v8h*)(ak + 3 * rstride);
        v8h hi3 = *(const v8h*)(ak + 3 * rstride + 16);
        v16h a_0 = __builtin_shufflevector(lo0, hi0, 0,1,2,3,4,5,6,7,8,9,10,11,12,13,14,15);
        v16h a_1 = __builtin_shufflevector(lo1, hi1, 0,1,2,3,4,5,6,7,8,9,10,11,12,13,14,15);
        v16h a_2 = __builtin_shufflevector(lo2, hi2, 0,1,2,3,4,5,6,7,8,9,10,11,12,13,14,15);
        v16h a_3 = __builtin_shufflevector(lo3, hi3, 0,1,2,3,4,5,6,7,8,9,10,11,12,13,14,15);
        acc0 = __builtin_amdgcn_wmma_f32_16x16x32_f16(false, a_0, false, b, (short)0, acc0, false, false);
        acc1 = __builtin_amdgcn_wmma_f32_16x16x32_f16(false, a_1, false, b, (short)0, acc1, false, false);
        acc2 = __builtin_amdgcn_wmma_f32_16x16x32_f16(false, a_2, false, b, (short)0, acc2, false, false);
        acc3 = __builtin_amdgcn_wmma_f32_16x16x32_f16(false, a_3, false, b, (short)0, acc3, false, false);
    }
    float bv = bias[c];
#pragma unroll
    for (int j = 0; j < 8; ++j) {
        int rr = row0 + j + (half << 3);
        Y[(size_t)rr * N + c]            = acc0[j] + bv;
        Y[(size_t)(rr + 16) * N + c]     = acc1[j] + bv;
        Y[(size_t)(rr + 32) * N + c]     = acc2[j] + bv;
        Y[(size_t)(rr + 48) * N + c]     = acc3[j] + bv;
    }
}

// ---------------------------------------------------------------------------
// Utility fills
// ---------------------------------------------------------------------------
__global__ void k_fill(float* p, long n, float v) {
    long i = (long)blockIdx.x * blockDim.x + threadIdx.x;
    if (i < n) p[i] = v;
}
__global__ void k_rows_bias(float* y, const float* __restrict__ bias, long rows, int D) {
    long i = (long)blockIdx.x * blockDim.x + threadIdx.x;
    if (i < rows * D) y[i] = bias[i % D];
}

// ---------------------------------------------------------------------------
// GATv2 edge kernels.  Edge e < NEDGE: (src,dst) from arrays; else self-loop.
// ---------------------------------------------------------------------------
__global__ void k_edge_logits(const int* __restrict__ es, const int* __restrict__ ed,
                              const float* __restrict__ XL, const float* __restrict__ XR,
                              const float* __restrict__ att,
                              float* __restrict__ elog, float* __restrict__ segmax,
                              int H, int C) {
    long idx = (long)blockIdx.x * blockDim.x + threadIdx.x;
    if (idx >= (long)EEDGE * H) return;
    int e = (int)(idx / H), h = (int)(idx % H);
    int s, d;
    if (e < NEDGE) { s = es[e]; d = ed[e]; } else { s = d = e - NEDGE; }
    int D = H * C;
    const float* xl = XL + (size_t)s * D + h * C;
    const float* xr = XR + (size_t)d * D + h * C;
    const float* at = att + (size_t)h * C;
    float acc = 0.f;
    for (int c0 = 0; c0 < C; ++c0) acc += at[c0] * lrelu(xl[c0] + xr[c0]);
    elog[(size_t)e * H + h] = acc;
    atomicMaxF(&segmax[(size_t)d * H + h], acc);
}

__global__ void k_edge_expsum(const int* __restrict__ ed, float* __restrict__ elog,
                              const float* __restrict__ segmax, float* __restrict__ segsum,
                              int H) {
    long idx = (long)blockIdx.x * blockDim.x + threadIdx.x;
    if (idx >= (long)EEDGE * H) return;
    int e = (int)(idx / H), h = (int)(idx % H);
    int d = (e < NEDGE) ? ed[e] : (e - NEDGE);
    float a = expf(elog[idx] - segmax[(size_t)d * H + h]);
    elog[idx] = a;
    atomicAdd(&segsum[(size_t)d * H + h], a);
}

__global__ void k_aggregate(const int* __restrict__ es, const int* __restrict__ ed,
                            const float* __restrict__ XL, const float* __restrict__ elog,
                            const float* __restrict__ segsum, float* __restrict__ out,
                            int H, int C) {
    long idx = (long)blockIdx.x * blockDim.x + threadIdx.x;
    if (idx >= (long)EEDGE * H) return;
    int e = (int)(idx / H), h = (int)(idx % H);
    int s, d;
    if (e < NEDGE) { s = es[e]; d = ed[e]; } else { s = d = e - NEDGE; }
    int D = H * C;
    float alpha = elog[(size_t)e * H + h] / segsum[(size_t)d * H + h];
    const float* xl = XL + (size_t)s * D + h * C;
    float* o = out + (size_t)d * D + h * C;
    for (int c0 = 0; c0 < C; ++c0) atomicAdd(&o[c0], alpha * xl[c0]);
}

// ---------------------------------------------------------------------------
// BatchNorm (training stats, biased var) + LeakyReLU; plain LeakyReLU
// ---------------------------------------------------------------------------
__global__ void k_bn_stats(const float* __restrict__ X, int rows, int D,
                           float* __restrict__ mean, float* __restrict__ rstd) {
    __shared__ float s1[256], s2[256];
    int d = blockIdx.x;
    float sum = 0.f, sq = 0.f;
    for (int r = threadIdx.x; r < rows; r += blockDim.x) {
        float v = X[(size_t)r * D + d];
        sum += v; sq += v * v;
    }
    s1[threadIdx.x] = sum; s2[threadIdx.x] = sq;
    __syncthreads();
    for (int s = 128; s > 0; s >>= 1) {
        if (threadIdx.x < s) { s1[threadIdx.x] += s1[threadIdx.x + s]; s2[threadIdx.x] += s2[threadIdx.x + s]; }
        __syncthreads();
    }
    if (threadIdx.x == 0) {
        float mu = s1[0] / rows;
        float var = s2[0] / rows - mu * mu;
        mean[d] = mu;
        rstd[d] = rsqrtf(var + 1e-5f);
    }
}
__global__ void k_bn_apply(float* X, long rows, int D,
                           const float* __restrict__ gamma, const float* __restrict__ beta,
                           const float* __restrict__ mean, const float* __restrict__ rstd) {
    long i = (long)blockIdx.x * blockDim.x + threadIdx.x;
    if (i >= rows * D) return;
    int d = (int)(i % D);
    float v = gamma[d] * (X[i] - mean[d]) * rstd[d] + beta[d];
    X[i] = lrelu(v);
}
__global__ void k_lrelu(float* X, long n) {
    long i = (long)blockIdx.x * blockDim.x + threadIdx.x;
    if (i < n) X[i] = lrelu(X[i]);
}

// ---------------------------------------------------------------------------
// Heads
// ---------------------------------------------------------------------------
__global__ void k_nmol_mlp(const float* __restrict__ px,
                           const float* W1, const float* b1,
                           const float* W2, const float* b2,
                           const float* W3, const float* b3,
                           float* __restrict__ logits) {
    int m = blockIdx.x * blockDim.x + threadIdx.x;
    if (m >= GNUM * NMOL) return;
    int g = m / NMOL, n = m % NMOL;
    const float* p = px + ((size_t)(g * NPG + n)) * 32;
    float h1[32], h2[16];
    for (int j = 0; j < 32; ++j) {
        float s = b1[j];
        for (int i = 0; i < 32; ++i) s += p[i] * W1[i * 32 + j];
        h1[j] = lrelu(s);
    }
    for (int j = 0; j < 16; ++j) {
        float s = b2[j];
        for (int i = 0; i < 32; ++i) s += h1[i] * W2[i * 16 + j];
        h2[j] = lrelu(s);
    }
    float s = b3[0];
    for (int i = 0; i < 16; ++i) s += h2[i] * W3[i];
    logits[m] = lrelu(s);
}

__global__ void k_graph_softmax(const float* __restrict__ logits, float* __restrict__ out, int R) {
    int g = blockIdx.x * blockDim.x + threadIdx.x;
    if (g >= GNUM) return;
    const float* l = logits + (size_t)g * R;
    float m = l[0];
    for (int i = 1; i < R; ++i) m = fmaxf(m, l[i]);
    float s = 0.f;
    for (int i = 0; i < R; ++i) s += expf(l[i] - m);
    float inv = 1.f / s;
    float* o = out + (size_t)g * R;
    for (int i = 0; i < R; ++i) o[i] = expf(l[i] - m) * inv;
}

__global__ void k_gmax(const float* __restrict__ v, int n, float* gm) {
    int i = blockIdx.x * blockDim.x + threadIdx.x;
    if (i < n) atomicMaxF(gm, v[i]);
}
__global__ void k_gexp(const float* __restrict__ v, int n, const float* gm,
                       float* __restrict__ tmp, float* gs) {
    int i = blockIdx.x * blockDim.x + threadIdx.x;
    if (i < n) { float a = expf(v[i] - *gm); tmp[i] = a; atomicAdd(gs, a); }
}
__global__ void k_gnorm(const float* __restrict__ tmp, int n, const float* gs,
                        float* __restrict__ out) {
    int i = blockIdx.x * blockDim.x + threadIdx.x;
    if (i < n) out[i] = tmp[i] / (*gs);
}

__global__ void k_nfull_mlp(const float* __restrict__ px, const float* __restrict__ nmol,
                            const float* W1, const float* b1,
                            const float* W2, const float* b2,
                            float* __restrict__ logits) {
    int node = blockIdx.x * blockDim.x + threadIdx.x;
    if (node >= NNODES) return;
    int g = node >> 6, n = node & 63;
    const float* p = px + (size_t)node * 32;
    float extra = (n < NMOL) ? nmol[g * NMOL + n] : -1.f;
    float h[16];
    for (int j = 0; j < 16; ++j) {
        float s = b1[j];
        for (int i = 0; i < 32; ++i) s += p[i] * W1[i * 16 + j];
        s += extra * W1[32 * 16 + j];
        h[j] = lrelu(s);
    }
    float s = b2[0];
    for (int i = 0; i < 16; ++i) s += h[i] * W2[i];
    logits[node] = lrelu(s);
}

__global__ void k_pool(const float* __restrict__ px, const float* __restrict__ nmglob,
                       const float* __restrict__ nfull, float* __restrict__ pbond) {
    int idx = blockIdx.x * blockDim.x + threadIdx.x;
    if (idx >= GNUM * 32) return;
    int g = idx >> 5, c = idx & 31;
    const float* base = px + (size_t)g * NPG * 32 + c;
    float sm = 0.f, snm = 0.f, snf = 0.f;
    for (int n = 0; n < NPG; ++n) {
        float v = base[(size_t)n * 32];
        sm += v;
        if (n < NMOL) snm += v * nmglob[g * NMOL + n];
        snf += v * nfull[g * NPG + n];
    }
    pbond[(size_t)g * 96 + c]      = sm / (float)NPG;
    pbond[(size_t)g * 96 + 32 + c] = snm;
    pbond[(size_t)g * 96 + 64 + c] = snf;
}

__global__ void k_b_head(const float* __restrict__ pbond,
                         const float* W1, const float* b1,
                         const float* W2, const float* b2,
                         float* __restrict__ bout) {
    int g = blockIdx.x * blockDim.x + threadIdx.x;
    if (g >= GNUM) return;
    const float* p = pbond + (size_t)g * 96;
    float h[16];
    for (int j = 0; j < 16; ++j) {
        float s = b1[j];
        for (int i = 0; i < 96; ++i) s += p[i] * W1[i * 16 + j];
        h[j] = lrelu(s);
    }
    float l[3];
    float m = -1e30f;
    for (int j = 0; j < 3; ++j) {
        float s = b2[j];
        for (int i = 0; i < 16; ++i) s += h[i] * W2[i * 3 + j];
        l[j] = lrelu(s);
        m = fmaxf(m, l[j]);
    }
    float sum = 0.f;
    for (int j = 0; j < 3; ++j) sum += expf(l[j] - m);
    for (int j = 0; j < 3; ++j) bout[g * 3 + j] = expf(l[j] - m) / sum;
}

__global__ void k_t_head(const float* __restrict__ pbond, const float* __restrict__ bout,
                         const float* __restrict__ nmol, const float* __restrict__ nfull,
                         const float* W, const float* b, float* __restrict__ tout) {
    int g = blockIdx.x * blockDim.x + threadIdx.x;
    if (g >= GNUM) return;
    float nm_mu = 0.f;
    for (int i = 0; i < NMOL; ++i) nm_mu += nmol[g * NMOL + i];
    nm_mu /= (float)NMOL;
    float nm_var = 0.f;
    for (int i = 0; i < NMOL; ++i) { float d = nmol[g * NMOL + i] - nm_mu; nm_var += d * d; }
    float nm_sd = sqrtf(nm_var / (float)(NMOL - 1));
    float nf_mu = 0.f;
    for (int i = 0; i < NPG; ++i) nf_mu += nfull[g * NPG + i];
    nf_mu /= (float)NPG;
    float nf_var = 0.f;
    for (int i = 0; i < NPG; ++i) { float d = nfull[g * NPG + i] - nf_mu; nf_var += d * d; }
    float nf_sd = sqrtf(nf_var / (float)(NPG - 1));
    float b_mu = (bout[g * 3] + bout[g * 3 + 1] + bout[g * 3 + 2]) / 3.f;
    float b_var = 0.f;
    for (int i = 0; i < 3; ++i) { float d = bout[g * 3 + i] - b_mu; b_var += d * d; }
    float b_sd = sqrtf(b_var / 2.f);
    float in[106];
    for (int i = 0; i < 96; ++i) in[i] = pbond[(size_t)g * 96 + i];
    for (int i = 0; i < 3; ++i) in[96 + i] = bout[g * 3 + i];
    in[99] = nm_mu; in[100] = nm_sd; in[101] = nf_mu; in[102] = nf_sd;
    in[103] = b_mu; in[104] = b_sd; in[105] = (float)NMOL;
    float l[2];
    for (int j = 0; j < 2; ++j) {
        float s = b[j];
        for (int i = 0; i < 106; ++i) s += in[i] * W[i * 2 + j];
        l[j] = lrelu(s);
    }
    float m = fmaxf(l[0], l[1]);
    float sum = expf(l[0] - m) + expf(l[1] - m);
    tout[g * 2 + 0] = expf(l[0] - m) / sum;
    tout[g * 2 + 1] = expf(l[1] - m) / sum;
}

__global__ void k_vpool(const float* __restrict__ vx, float* __restrict__ vpool) {
    int idx = blockIdx.x * blockDim.x + threadIdx.x;
    if (idx >= GNUM * 32) return;
    int g = idx >> 5, c = idx & 31;
    const float* base = vx + (size_t)g * NPG * 32 + c;
    float s = 0.f;
    for (int n = 0; n < NPG; ++n) s += base[(size_t)n * 32];
    vpool[idx] = s / (float)NPG;
}

__global__ void k_v_head(const float* __restrict__ vpool,
                         const float* W1, const float* b1,
                         const float* W2, const float* b2,
                         float* __restrict__ vout) {
    int g = blockIdx.x * blockDim.x + threadIdx.x;
    if (g >= GNUM) return;
    const float* p = vpool + (size_t)g * 32;
    float h[16];
    for (int j = 0; j < 16; ++j) {
        float s = b1[j];
        for (int i = 0; i < 32; ++i) s += p[i] * W1[i * 16 + j];
        h[j] = lrelu(s);
    }
    float s = b2[0];
    for (int i = 0; i < 16; ++i) s += h[i] * W2[i];
    vout[g] = lrelu(s);
}

// ---------------------------------------------------------------------------
// Host orchestration
// ---------------------------------------------------------------------------
static inline int gridl(long n) { return (int)((n + 255) / 256); }

static void gat_layer(const float* X, int Cin, int H, int C,
                      const float* Wl, const float* bl, const float* Wr, const float* br,
                      const float* att, const float* bias,
                      float* out, float* XLp, float* XRp,
                      _Float16* Xh, _Float16* WpL, _Float16* WpR,
                      float* elog, float* segmax, float* segsum,
                      const int* es, const int* ed, hipStream_t stream) {
    int D = H * C;
    int Kp = (Cin + 31) & ~31;
    // stage inputs in f16 (padded) + pack weights into WMMA fragment order
    k_cvt_pad_f16<<<gridl((long)NNODES * Kp), 256, 0, stream>>>(X, Xh, NNODES, Cin, Kp);
    int wtot = (Kp / 32) * (D / 16) * 512;
    k_pack_w<<<gridl(wtot), 256, 0, stream>>>(Wl, WpL, Cin, Kp, D);
    k_pack_w<<<gridl(wtot), 256, 0, stream>>>(Wr, WpR, Cin, Kp, D);
    dim3 gg(NNODES / 256, D / 16);   // 4 waves/block, 64 rows per wave
    k_gemm_wmma<<<gg, 128, 0, stream>>>(Xh, WpL, bl, XLp, NNODES, Kp, D);
    k_gemm_wmma<<<gg, 128, 0, stream>>>(Xh, WpR, br, XRp, NNODES, Kp, D);
    long nh = (long)NNODES * H;
    k_fill<<<gridl(nh), 256, 0, stream>>>(segmax, nh, -__builtin_inff());
    k_fill<<<gridl(nh), 256, 0, stream>>>(segsum, nh, 0.f);
    long eh = (long)EEDGE * H;
    k_edge_logits<<<gridl(eh), 256, 0, stream>>>(es, ed, XLp, XRp, att, elog, segmax, H, C);
    k_edge_expsum<<<gridl(eh), 256, 0, stream>>>(ed, elog, segmax, segsum, H);
    long nd = (long)NNODES * D;
    k_rows_bias<<<gridl(nd), 256, 0, stream>>>(out, bias, NNODES, D);
    k_aggregate<<<gridl(eh), 256, 0, stream>>>(es, ed, XLp, elog, segsum, out, H, C);
}

extern "C" void kernel_launch(void* const* d_in, const int* in_sizes, int n_in,
                              void* d_out, int out_size, void* d_ws, size_t ws_size,
                              hipStream_t stream) {
    (void)in_sizes; (void)n_in; (void)out_size; (void)ws_size;
    // Input order: JAX pytree flatten, dict keys sorted ('W'<'b', 'Wl'<'Wr'<'att'<'bias'<'bl'<'br', 'beta'<'gamma')
    const float* x = (const float*)d_in[0];
    const float* b1W = (const float*)d_in[1];  const float* b1b = (const float*)d_in[2];
    const float* b2W = (const float*)d_in[3];  const float* b2b = (const float*)d_in[4];
    const float* nf1W = (const float*)d_in[5]; const float* nf1b = (const float*)d_in[6];
    const float* nf2W = (const float*)d_in[7]; const float* nf2b = (const float*)d_in[8];
    const float* nm1W = (const float*)d_in[9]; const float* nm1b = (const float*)d_in[10];
    const float* nm2W = (const float*)d_in[11]; const float* nm2b = (const float*)d_in[12];
    const float* nm3W = (const float*)d_in[13]; const float* nm3b = (const float*)d_in[14];
    const float* bn1be = (const float*)d_in[15]; const float* bn1ga = (const float*)d_in[16];
    const float* bn2be = (const float*)d_in[17]; const float* bn2ga = (const float*)d_in[18];
    const float* bn3be = (const float*)d_in[19]; const float* bn3ga = (const float*)d_in[20];
    const float* pc1Wl = (const float*)d_in[21]; const float* pc1Wr = (const float*)d_in[22];
    const float* pc1at = (const float*)d_in[23]; const float* pc1bi = (const float*)d_in[24];
    const float* pc1bl = (const float*)d_in[25]; const float* pc1br = (const float*)d_in[26];
    const float* pc2Wl = (const float*)d_in[27]; const float* pc2Wr = (const float*)d_in[28];
    const float* pc2at = (const float*)d_in[29]; const float* pc2bi = (const float*)d_in[30];
    const float* pc2bl = (const float*)d_in[31]; const float* pc2br = (const float*)d_in[32];
    const float* pc3Wl = (const float*)d_in[33]; const float* pc3Wr = (const float*)d_in[34];
    const float* pc3at = (const float*)d_in[35]; const float* pc3bi = (const float*)d_in[36];
    const float* pc3bl = (const float*)d_in[37]; const float* pc3br = (const float*)d_in[38];
    const float* t1W = (const float*)d_in[39]; const float* t1b = (const float*)d_in[40];
    const float* v1W = (const float*)d_in[41]; const float* v1b = (const float*)d_in[42];
    const float* v2W = (const float*)d_in[43]; const float* v2b = (const float*)d_in[44];
    const float* vc1Wl = (const float*)d_in[45]; const float* vc1Wr = (const float*)d_in[46];
    const float* vc1at = (const float*)d_in[47]; const float* vc1bi = (const float*)d_in[48];
    const float* vc1bl = (const float*)d_in[49]; const float* vc1br = (const float*)d_in[50];
    const float* vc2Wl = (const float*)d_in[51]; const float* vc2Wr = (const float*)d_in[52];
    const float* vc2at = (const float*)d_in[53]; const float* vc2bi = (const float*)d_in[54];
    const float* vc2bl = (const float*)d_in[55]; const float* vc2br = (const float*)d_in[56];
    const float* vc3Wl = (const float*)d_in[57]; const float* vc3Wr = (const float*)d_in[58];
    const float* vc3at = (const float*)d_in[59]; const float* vc3bi = (const float*)d_in[60];
    const float* vc3bl = (const float*)d_in[61]; const float* vc3br = (const float*)d_in[62];
    const int* esrc = (const int*)d_in[63];
    const int* edst = (const int*)d_in[64];

    // Output layout: t(2048) | nmol(55296) | nfull(65536) | b(3072) | v(1024)
    float* out_t = (float*)d_out;
    float* out_nmol = out_t + 2048;
    float* out_nfull = out_nmol + GNUM * NMOL;
    float* out_b = out_nfull + NNODES;
    float* out_v = out_b + GNUM * 3;

    // Workspace carve-up (float units)
    float* ws = (float*)d_ws;
    size_t o = 0;
    float* nodeA  = ws + o; o += (size_t)NNODES * 192;
    float* XLp    = ws + o; o += (size_t)NNODES * 192;
    float* XRp    = ws + o; o += (size_t)NNODES * 192;
    float* elog   = ws + o; o += (size_t)EEDGE * 3;
    float* segmax = ws + o; o += (size_t)NNODES * 3;
    float* segsum = ws + o; o += (size_t)NNODES * 3;
    _Float16* Xh  = (_Float16*)(ws + o); o += (size_t)NNODES * 96;   // 65536 x 192 halves
    _Float16* WpL = (_Float16*)(ws + o); o += 8192;                  // 16384 halves
    _Float16* WpR = (_Float16*)(ws + o); o += 8192;
    float* bmean  = ws + o; o += 192;
    float* brstd  = ws + o; o += 192;
    float* nmlog  = ws + o; o += GNUM * NMOL;
    float* nmtmp  = ws + o; o += GNUM * NMOL;
    float* nmglob = ws + o; o += GNUM * NMOL;
    float* nflog  = ws + o; o += NNODES;
    float* pbond  = ws + o; o += (size_t)GNUM * 96;
    float* vpool  = ws + o; o += (size_t)GNUM * 32;
    float* gscal  = ws + o; o += 2;   // [gmax, gsum]

    // ---- policy GNN ----
    gat_layer(x, 24, 3, 64, pc1Wl, pc1bl, pc1Wr, pc1br, pc1at, pc1bi,
              nodeA, XLp, XRp, Xh, WpL, WpR, elog, segmax, segsum, esrc, edst, stream);
    k_bn_stats<<<192, 256, 0, stream>>>(nodeA, NNODES, 192, bmean, brstd);
    k_bn_apply<<<gridl((long)NNODES * 192), 256, 0, stream>>>(nodeA, NNODES, 192, bn1ga, bn1be, bmean, brstd);

    gat_layer(nodeA, 192, 1, 64, pc2Wl, pc2bl, pc2Wr, pc2br, pc2at, pc2bi,
              nodeA, XLp, XRp, Xh, WpL, WpR, elog, segmax, segsum, esrc, edst, stream);
    k_bn_stats<<<64, 256, 0, stream>>>(nodeA, NNODES, 64, bmean, brstd);
    k_bn_apply<<<gridl((long)NNODES * 64), 256, 0, stream>>>(nodeA, NNODES, 64, bn2ga, bn2be, bmean, brstd);

    gat_layer(nodeA, 64, 1, 32, pc3Wl, pc3bl, pc3Wr, pc3br, pc3at, pc3bi,
              nodeA, XLp, XRp, Xh, WpL, WpR, elog, segmax, segsum, esrc, edst, stream);
    k_bn_stats<<<32, 256, 0, stream>>>(nodeA, NNODES, 32, bmean, brstd);
    k_bn_apply<<<gridl((long)NNODES * 32), 256, 0, stream>>>(nodeA, NNODES, 32, bn3ga, bn3be, bmean, brstd);
    // nodeA now holds p_x [65536, 32]

    // ---- nmol head ----
    k_nmol_mlp<<<gridl(GNUM * NMOL), 256, 0, stream>>>(nodeA, nm1W, nm1b, nm2W, nm2b, nm3W, nm3b, nmlog);
    k_graph_softmax<<<gridl(GNUM), 256, 0, stream>>>(nmlog, out_nmol, NMOL);

    // nmol_glob: global softmax over all G*NM rows
    k_fill<<<1, 256, 0, stream>>>(gscal, 1, -__builtin_inff());
    k_fill<<<1, 256, 0, stream>>>(gscal + 1, 1, 0.f);
    k_gmax<<<gridl(GNUM * NMOL), 256, 0, stream>>>(out_nmol, GNUM * NMOL, gscal);
    k_gexp<<<gridl(GNUM * NMOL), 256, 0, stream>>>(out_nmol, GNUM * NMOL, gscal, nmtmp, gscal + 1);
    k_gnorm<<<gridl(GNUM * NMOL), 256, 0, stream>>>(nmtmp, GNUM * NMOL, gscal + 1, nmglob);

    // ---- nfull head ----
    k_nfull_mlp<<<gridl(NNODES), 256, 0, stream>>>(nodeA, out_nmol, nf1W, nf1b, nf2W, nf2b, nflog);
    k_graph_softmax<<<gridl(GNUM), 256, 0, stream>>>(nflog, out_nfull, NPG);

    // ---- pooling + bond head ----
    k_pool<<<gridl(GNUM * 32), 256, 0, stream>>>(nodeA, nmglob, out_nfull, pbond);
    k_b_head<<<gridl(GNUM), 256, 0, stream>>>(pbond, b1W, b1b, b2W, b2b, out_b);

    // ---- termination head ----
    k_t_head<<<gridl(GNUM), 256, 0, stream>>>(pbond, out_b, out_nmol, out_nfull, t1W, t1b, out_t);

    // ---- value GNN (reuses big buffers; policy heads already consumed p_x) ----
    gat_layer(x, 24, 1, 64, vc1Wl, vc1bl, vc1Wr, vc1br, vc1at, vc1bi,
              nodeA, XLp, XRp, Xh, WpL, WpR, elog, segmax, segsum, esrc, edst, stream);
    k_lrelu<<<gridl((long)NNODES * 64), 256, 0, stream>>>(nodeA, (long)NNODES * 64);
    gat_layer(nodeA, 64, 1, 64, vc2Wl, vc2bl, vc2Wr, vc2br, vc2at, vc2bi,
              nodeA, XLp, XRp, Xh, WpL, WpR, elog, segmax, segsum, esrc, edst, stream);
    k_lrelu<<<gridl((long)NNODES * 64), 256, 0, stream>>>(nodeA, (long)NNODES * 64);
    gat_layer(nodeA, 64, 1, 32, vc3Wl, vc3bl, vc3Wr, vc3br, vc3at, vc3bi,
              nodeA, XLp, XRp, Xh, WpL, WpR, elog, segmax, segsum, esrc, edst, stream);
    k_lrelu<<<gridl((long)NNODES * 32), 256, 0, stream>>>(nodeA, (long)NNODES * 32);

    k_vpool<<<gridl(GNUM * 32), 256, 0, stream>>>(nodeA, vpool);
    k_v_head<<<gridl(GNUM), 256, 0, stream>>>(vpool, v1W, v1b, v2W, v2b, out_v);
}
